// GCN_normal_47218870452450
// MI455X (gfx1250) — matrix operand order repaired and checked by reading, not compiled, not run on hardware
//
#include <hip/hip_runtime.h>
#include <hip/hip_bf16.h>

typedef __attribute__((ext_vector_type(16))) _Float16 v16h;
typedef __attribute__((ext_vector_type(8)))  _Float16 v8h;
typedef __attribute__((ext_vector_type(8)))  float    v8f;

#define N_NODES 8192
#define NFEAT   500
#define KPAD    512
#define NHID    256

// ---------------------------------------------------------------------------
// WMMA helpers (CDNA5 v_wmma_f32_16x16x32_f16, wave32)
// ---------------------------------------------------------------------------
__device__ __forceinline__ v8f wmma_f16(v16h a, v16h b, v8f c) {
  // 8 args: (neg_a, A, neg_b, B, c_mod, C, reuse_a, reuse_b)
  return __builtin_amdgcn_wmma_f32_16x16x32_f16(false, a, false, b, (short)0, c,
                                                false, false);
}

// A-fragment (16x32, M x K) from a row-major matrix L[M][ldk] (f16).
// Lane l (<16): rows M=l, K 0..15; lanes >=16: M=l-16,
// K 16..31.  Two contiguous 16-byte loads per lane.
__device__ __forceinline__ v16h load_fragA(const _Float16* L, int ldk, int row,
                                           int kbase, int lane) {
  const _Float16* p = L + (size_t)row * ldk + kbase + ((lane >> 4) << 4);
  union { v16h v; v8h h[2]; } u;
  u.h[0] = *(const v8h*)(p);
  u.h[1] = *(const v8h*)(p + 8);
  return u.v;
}

// B-fragment (32x16, K x N) from the TRANSPOSED right matrix Rt[N][ldk] (f16):
// Rt[n][k] = R[k][n].  Lane l: column n=l&15, K-range (l>>4)*16 .. +15 ->
// 16 contiguous halves = one 32-byte load per lane.
__device__ __forceinline__ v16h load_fragB(const _Float16* Rt, int ldk, int col,
                                           int kbase, int lane) {
  const _Float16* p = Rt + (size_t)col * ldk + kbase + ((lane >> 4) << 4);
  return *(const v16h*)(p);
}

// ---------------------------------------------------------------------------
// Prep kernels
// ---------------------------------------------------------------------------
__global__ void init_kernel(int* deg, int* maxbits, float* out) {
  int i = blockIdx.x * blockDim.x + threadIdx.x;
  if (i < N_NODES) deg[i] = 0;
  if (i < 2 * NHID) out[i] = 0.f;
  if (i == 0) *maxbits = 0;
}

// X f32 [8192][500] -> Xh f16 [8192][512] (zero-padded K)
__global__ void cvt_x_kernel(const float* __restrict__ X, _Float16* __restrict__ Xh) {
  int idx = blockIdx.x * blockDim.x + threadIdx.x;  // over 8192*512
  int row = idx >> 9, k = idx & (KPAD - 1);
  Xh[idx] = (k < NFEAT) ? (_Float16)X[(size_t)row * NFEAT + k] : (_Float16)0.f;
}

// W f32 [K][N] -> Wt f16 [N][Kpad] (transposed, zero-padded K)
__global__ void cvt_wt_kernel(const float* __restrict__ W, _Float16* __restrict__ Wt,
                              int K, int N, int Kpad) {
  int idx = blockIdx.x * blockDim.x + threadIdx.x;  // over N*Kpad
  int n = idx / Kpad, k = idx - n * Kpad;
  Wt[idx] = (k < K) ? (_Float16)W[(size_t)k * N + n] : (_Float16)0.f;
}

// sq[i] = ||x_i||^2 (full f32 precision from the original features)
__global__ void sqnorm_kernel(const float* __restrict__ X, float* __restrict__ sq) {
  int lane = threadIdx.x & 31;
  int row = blockIdx.x * 8 + (threadIdx.x >> 5);
  float s = 0.f;
  for (int k = lane; k < NFEAT; k += 32) {
    float v = X[(size_t)row * NFEAT + k];
    s += v * v;
  }
  for (int off = 16; off; off >>= 1) s += __shfl_xor(s, off, 32);
  if (lane == 0) sq[row] = s;
}

// ---------------------------------------------------------------------------
// Gram + pairwise distance:  d[i][j] = sq[i] + sq[j] - 2 * <x_i, x_j>
// Fuses the global max (for the threshold) via wave-reduce + int atomicMax
// (valid: d >= 0 so the IEEE bit pattern is monotone as a signed int).
// Block = 8 waves tiling 128x64; each wave a 32x32 register tile (4 WMMAs/K).
// ---------------------------------------------------------------------------
__global__ void __launch_bounds__(256)
gram_d_kernel(const _Float16* __restrict__ Xh, const float* __restrict__ sq,
              float* __restrict__ dmat, int* __restrict__ maxbits) {
  const int lane = threadIdx.x & 31;
  const int wave = threadIdx.x >> 5;
  const int rowBase = blockIdx.x * 128 + (wave & 3) * 32;
  const int colBase = blockIdx.y * 64 + (wave >> 2) * 32;
  const int r0 = rowBase + (lane & 15), r1 = r0 + 16;
  const int c0 = colBase + (lane & 15), c1 = c0 + 16;
  v8f acc00 = {}, acc01 = {}, acc10 = {}, acc11 = {};
  for (int k = 0; k < KPAD; k += 32) {
    v16h a0 = load_fragA(Xh, KPAD, r0, k, lane);
    v16h a1 = load_fragA(Xh, KPAD, r1, k, lane);
    v16h b0 = load_fragB(Xh, KPAD, c0, k, lane);  // gram: Rt == Xh
    v16h b1 = load_fragB(Xh, KPAD, c1, k, lane);
    acc00 = wmma_f16(a0, b0, acc00);
    acc01 = wmma_f16(a0, b1, acc01);
    acc10 = wmma_f16(a1, b0, acc10);
    acc11 = wmma_f16(a1, b1, acc11);
  }
  // D layout: element r of the accumulator is M = rowBase + 8*(lane>>4) + r,
  // N = colBase + (lane&15).
  const int n0 = colBase + (lane & 15), n1 = n0 + 16;
  const int mb = rowBase + ((lane >> 4) << 3);
  const float sn0 = sq[n0], sn1 = sq[n1];
  float wmax = 0.f;
  for (int r = 0; r < 8; ++r) {
    int m0 = mb + r, m1 = m0 + 16;
    float sm0 = sq[m0], sm1 = sq[m1];
    float d00 = sm0 + sn0 - 2.f * acc00[r];
    float d01 = sm0 + sn1 - 2.f * acc01[r];
    float d10 = sm1 + sn0 - 2.f * acc10[r];
    float d11 = sm1 + sn1 - 2.f * acc11[r];
    dmat[(size_t)m0 * N_NODES + n0] = d00;
    dmat[(size_t)m0 * N_NODES + n1] = d01;
    dmat[(size_t)m1 * N_NODES + n0] = d10;
    dmat[(size_t)m1 * N_NODES + n1] = d11;
    wmax = fmaxf(wmax, fmaxf(fmaxf(d00, d01), fmaxf(d10, d11)));
  }
  for (int off = 16; off; off >>= 1)
    wmax = fmaxf(wmax, __shfl_xor(wmax, off, 32));
  if (lane == 0) atomicMax(maxbits, __float_as_int(wmax));
}

// deg[j] = #{ i < j : d[i][j] < t }   (self-loop added later as +1)
__global__ void deg_kernel(const float* __restrict__ dmat,
                           const int* __restrict__ maxbits, int* __restrict__ deg) {
  int j = blockIdx.x * blockDim.x + threadIdx.x;
  float t = 0.5f * __int_as_float(*maxbits);
  int i0 = blockIdx.y * 512;
  int iend = i0 + 512;
  if (iend > j) iend = j;
  int cnt = 0;
  for (int i = i0; i < iend; ++i)
    if (dmat[(size_t)i * N_NODES + j] < t) cnt++;
  if (cnt) atomicAdd(deg + j, cnt);
}

__global__ void dinv_kernel(const int* __restrict__ deg, float* __restrict__ dinv) {
  int j = blockIdx.x * blockDim.x + threadIdx.x;
  dinv[j] = rsqrtf((float)deg[j] + 1.0f);
}

// A[j][i] = dinv_i * dinv_j * M[i,j]  (row-major f16; reads d[j][i] by symmetry
// for coalesced access)
__global__ void build_a_kernel(const float* __restrict__ dmat,
                               const float* __restrict__ dinv,
                               const int* __restrict__ maxbits,
                               _Float16* __restrict__ A) {
  int i = blockIdx.x * blockDim.x + threadIdx.x;
  int j = blockIdx.y;
  float t = 0.5f * __int_as_float(*maxbits);
  float v = 0.f;
  if (i == j) {
    float di = dinv[i];
    v = di * di;
  } else if (i < j && dmat[(size_t)j * N_NODES + i] < t) {
    v = dinv[i] * dinv[j];
  }
  A[(size_t)j * N_NODES + i] = (_Float16)v;
}

// ---------------------------------------------------------------------------
// GEMM 1: Pt = (L @ W)^T.  L row-major [8192][ldk] f16, Wt = W^T [256][ldk] f16.
// Writes transposed f16 so it can feed the aggregation GEMM's B-fragments
// (transposed store is the *contiguous* direction of the D layout).
// ---------------------------------------------------------------------------
__global__ void __launch_bounds__(256)
gemm_pt_kernel(const _Float16* __restrict__ L, const _Float16* __restrict__ Wt,
               _Float16* __restrict__ Pt, int ldk) {
  const int lane = threadIdx.x & 31;
  const int wave = threadIdx.x >> 5;
  const int rowBase = blockIdx.x * 128 + (wave & 3) * 32;
  const int colBase = blockIdx.y * 64 + (wave >> 2) * 32;
  const int r0 = rowBase + (lane & 15), r1 = r0 + 16;
  const int c0 = colBase + (lane & 15), c1 = c0 + 16;
  v8f acc00 = {}, acc01 = {}, acc10 = {}, acc11 = {};
  for (int k = 0; k < ldk; k += 32) {
    v16h a0 = load_fragA(L, ldk, r0, k, lane);
    v16h a1 = load_fragA(L, ldk, r1, k, lane);
    v16h b0 = load_fragB(Wt, ldk, c0, k, lane);
    v16h b1 = load_fragB(Wt, ldk, c1, k, lane);
    acc00 = wmma_f16(a0, b0, acc00);
    acc01 = wmma_f16(a0, b1, acc01);
    acc10 = wmma_f16(a1, b0, acc10);
    acc11 = wmma_f16(a1, b1, acc11);
  }
  const int n0 = colBase + (lane & 15), n1 = n0 + 16;
  const int mb = rowBase + ((lane >> 4) << 3);
  v8h h00, h01, h10, h11;
  for (int r = 0; r < 8; ++r) {
    h00[r] = (_Float16)acc00[r];
    h01[r] = (_Float16)acc01[r];
    h10[r] = (_Float16)acc10[r];
    h11[r] = (_Float16)acc11[r];
  }
  *(v8h*)(Pt + (size_t)n0 * N_NODES + mb)      = h00;
  *(v8h*)(Pt + (size_t)n1 * N_NODES + mb)      = h01;
  *(v8h*)(Pt + (size_t)n0 * N_NODES + mb + 16) = h10;
  *(v8h*)(Pt + (size_t)n1 * N_NODES + mb + 16) = h11;
}

// ---------------------------------------------------------------------------
// GEMM 2 (aggregation): H = relu(A @ P + b), A [8192][8192] f16 row-major,
// Pt = P^T [256][8192] f16.  H written row-major f16 (next layer's L operand).
// ---------------------------------------------------------------------------
__global__ void __launch_bounds__(256)
gemm_agg_kernel(const _Float16* __restrict__ A, const _Float16* __restrict__ Pt,
                const float* __restrict__ bias, _Float16* __restrict__ H) {
  const int lane = threadIdx.x & 31;
  const int wave = threadIdx.x >> 5;
  const int rowBase = blockIdx.x * 128 + (wave & 3) * 32;
  const int colBase = blockIdx.y * 64 + (wave >> 2) * 32;
  const int r0 = rowBase + (lane & 15), r1 = r0 + 16;
  const int c0 = colBase + (lane & 15), c1 = c0 + 16;
  v8f acc00 = {}, acc01 = {}, acc10 = {}, acc11 = {};
  for (int k = 0; k < N_NODES; k += 32) {
    v16h a0 = load_fragA(A, N_NODES, r0, k, lane);
    v16h a1 = load_fragA(A, N_NODES, r1, k, lane);
    v16h b0 = load_fragB(Pt, N_NODES, c0, k, lane);
    v16h b1 = load_fragB(Pt, N_NODES, c1, k, lane);
    acc00 = wmma_f16(a0, b0, acc00);
    acc01 = wmma_f16(a0, b1, acc01);
    acc10 = wmma_f16(a1, b0, acc10);
    acc11 = wmma_f16(a1, b1, acc11);
  }
  const int n0 = colBase + (lane & 15), n1 = n0 + 16;
  const int mb = rowBase + ((lane >> 4) << 3);
  const float b0v = bias[n0], b1v = bias[n1];
  for (int r = 0; r < 8; ++r) {
    int m0 = mb + r, m1 = m0 + 16;
    H[(size_t)m0 * NHID + n0] = (_Float16)fmaxf(acc00[r] + b0v, 0.f);
    H[(size_t)m0 * NHID + n1] = (_Float16)fmaxf(acc01[r] + b1v, 0.f);
    H[(size_t)m1 * NHID + n0] = (_Float16)fmaxf(acc10[r] + b0v, 0.f);
    H[(size_t)m1 * NHID + n1] = (_Float16)fmaxf(acc11[r] + b1v, 0.f);
  }
}

// Column mean of H, accumulated into out[n] and out[256+n] (concat(p,p)).
__global__ void pool_kernel(const _Float16* __restrict__ H, float* __restrict__ out) {
  int n = threadIdx.x;           // 256 threads = one per column
  int r0 = blockIdx.x * 128;     // 64 blocks x 128 rows
  float s = 0.f;
  for (int i = 0; i < 128; ++i) s += (float)H[(size_t)(r0 + i) * NHID + n];
  s *= (1.0f / (float)N_NODES);
  atomicAdd(out + n, s);
  atomicAdd(out + NHID + n, s);
}

// ---------------------------------------------------------------------------
// Host launcher
// ---------------------------------------------------------------------------
extern "C" void kernel_launch(void* const* d_in, const int* in_sizes, int n_in,
                              void* d_out, int out_size, void* d_ws, size_t ws_size,
                              hipStream_t stream) {
  const float* X  = (const float*)d_in[0];
  const float* W1 = (const float*)d_in[1];
  const float* b1 = (const float*)d_in[2];
  const float* W2 = (const float*)d_in[3];
  const float* b2 = (const float*)d_in[4];
  const float* W3 = (const float*)d_in[5];
  const float* b3 = (const float*)d_in[6];
  float* out = (float*)d_out;

  char* ws = (char*)d_ws;
  size_t off = 0;
  float*    dmat = (float*)(ws + off);    off += (size_t)N_NODES * N_NODES * 4; // 256MB
  _Float16* A    = (_Float16*)(ws + off); off += (size_t)N_NODES * N_NODES * 2; // 128MB
  _Float16* Xh   = (_Float16*)(ws + off); off += (size_t)N_NODES * KPAD * 2;
  _Float16* W1t  = (_Float16*)(ws + off); off += (size_t)NHID * KPAD * 2;
  _Float16* W2t  = (_Float16*)(ws + off); off += (size_t)NHID * NHID * 2;
  _Float16* W3t  = (_Float16*)(ws + off); off += (size_t)NHID * NHID * 2;
  _Float16* Pt   = (_Float16*)(ws + off); off += (size_t)NHID * N_NODES * 2;
  _Float16* H    = (_Float16*)(ws + off); off += (size_t)N_NODES * NHID * 2;
  float*    sq   = (float*)(ws + off);    off += (size_t)N_NODES * 4;
  float*    dinv = (float*)(ws + off);    off += (size_t)N_NODES * 4;
  int*      deg  = (int*)(ws + off);      off += (size_t)N_NODES * 4;
  int*      maxb = (int*)(ws + off);      off += 16;

  init_kernel<<<32, 256, 0, stream>>>(deg, maxb, out);
  cvt_x_kernel<<<(N_NODES * KPAD) / 256, 256, 0, stream>>>(X, Xh);
  cvt_wt_kernel<<<(NHID * KPAD) / 256, 256, 0, stream>>>(W1, W1t, NFEAT, NHID, KPAD);
  cvt_wt_kernel<<<(NHID * NHID) / 256, 256, 0, stream>>>(W2, W2t, NHID, NHID, NHID);
  cvt_wt_kernel<<<(NHID * NHID) / 256, 256, 0, stream>>>(W3, W3t, NHID, NHID, NHID);
  sqnorm_kernel<<<N_NODES / 8, 256, 0, stream>>>(X, sq);

  // adjacency construction
  gram_d_kernel<<<dim3(N_NODES / 128, N_NODES / 64), 256, 0, stream>>>(Xh, sq, dmat, maxb);
  deg_kernel<<<dim3(N_NODES / 256, 16), 256, 0, stream>>>(dmat, maxb, deg);
  dinv_kernel<<<N_NODES / 256, 256, 0, stream>>>(deg, dinv);
  build_a_kernel<<<dim3(N_NODES / 256, N_NODES), 256, 0, stream>>>(dmat, dinv, maxb, A);

  // layer 1
  gemm_pt_kernel<<<dim3(N_NODES / 128, NHID / 64), 256, 0, stream>>>(Xh, W1t, Pt, KPAD);
  gemm_agg_kernel<<<dim3(N_NODES / 128, NHID / 64), 256, 0, stream>>>(A, Pt, b1, H);
  pool_kernel<<<N_NODES / 128, 256, 0, stream>>>(H, out);
  // layer 2
  gemm_pt_kernel<<<dim3(N_NODES / 128, NHID / 64), 256, 0, stream>>>(H, W2t, Pt, NHID);
  gemm_agg_kernel<<<dim3(N_NODES / 128, NHID / 64), 256, 0, stream>>>(A, Pt, b2, H);
  pool_kernel<<<N_NODES / 128, 256, 0, stream>>>(H, out);
  // layer 3
  gemm_pt_kernel<<<dim3(N_NODES / 128, NHID / 64), 256, 0, stream>>>(H, W3t, Pt, NHID);
  gemm_agg_kernel<<<dim3(N_NODES / 128, NHID / 64), 256, 0, stream>>>(A, Pt, b3, H);
  pool_kernel<<<N_NODES / 128, 256, 0, stream>>>(H, out);
}
